// TSPGNN_79456894976559
// MI455X (gfx1250) — compile-verified, matrix operand order checked by reading, not run on hardware
//
#include <hip/hip_runtime.h>
#include <hip/hip_bf16.h>

typedef __attribute__((ext_vector_type(16))) __bf16 v16bf;
typedef __attribute__((ext_vector_type(8)))  __bf16 v8bf;
typedef __attribute__((ext_vector_type(8)))  float  v8f;

#define CAT16(lo, hi) __builtin_shufflevector(lo, hi, 0, 1, 2, 3, 4, 5, 6, 7, 8, 9, 10, 11, 12, 13, 14, 15)

// ---------------------------------------------------------------- utilities

__global__ void tsp_fill_kernel(float* __restrict__ p, float v, long long n) {
    long long t = (long long)blockIdx.x * blockDim.x + threadIdx.x;
    if (t < n) p[t] = v;
}

// deg[col[e]] += 1  (deg pre-filled with 1.0 for the self loop)
__global__ void tsp_deg_kernel(const long long* __restrict__ col,
                               float* __restrict__ deg, int E) {
    int e = blockIdx.x * blockDim.x + threadIdx.x;
    if (e < E) atomicAdd(deg + (int)col[e], 1.0f);
}

// deg -> deg^{-1/2} in place (deg >= 1 always)
__global__ void tsp_rsqrt_kernel(float* __restrict__ deg, int N) {
    int i = blockIdx.x * blockDim.x + threadIdx.x;
    if (i < N) deg[i] = rsqrtf(deg[i]);
}

// s1[i][j] = dinv[i] * (x[i][0]*W1[0][j] + x[i][1]*W1[1][j]);  F=64
__global__ void tsp_s1_kernel(const float* __restrict__ x,
                              const float* __restrict__ W1,
                              const float* __restrict__ dinv,
                              float* __restrict__ s1, int N) {
    int t = blockIdx.x * blockDim.x + threadIdx.x;
    if (t >= N * 64) return;
    int i = t >> 6, j = t & 63;
    float v = fmaf(x[2 * i], W1[j], x[2 * i + 1] * W1[64 + j]);
    s1[t] = dinv[i] * v;
}

// agg[col[e]][c..c+3] += s[row[e]][c..c+3]
__global__ void tsp_scatter_kernel(const float* __restrict__ s,
                                   const long long* __restrict__ row,
                                   const long long* __restrict__ col,
                                   float* __restrict__ agg, int E, int F) {
    int chunks = F >> 2;
    long long t = (long long)blockIdx.x * blockDim.x + threadIdx.x;
    if (t >= (long long)E * chunks) return;
    int e = (int)(t / chunks);
    int c = ((int)(t % chunks)) << 2;
    int src = (int)row[e];
    int dst = (int)col[e];
    float4 v = *(const float4*)(s + (long long)src * F + c);
    float* ap = agg + (long long)dst * F + c;
    atomicAdd(ap + 0, v.x);
    atomicAdd(ap + 1, v.y);
    atomicAdd(ap + 2, v.z);
    atomicAdd(ap + 3, v.w);
}

// h[i][j] = bf16(relu(dinv[i]*(agg[i][j] + s[i][j]) + bias[j]))
__global__ void tsp_finalize_bf16_kernel(const float* __restrict__ s,
                                         const float* __restrict__ agg,
                                         const float* __restrict__ dinv,
                                         const float* __restrict__ bias,
                                         __bf16* __restrict__ h, int N, int F) {
    long long t = (long long)blockIdx.x * blockDim.x + threadIdx.x;
    if (t >= (long long)N * F) return;
    int i = (int)(t / F);
    int j = (int)(t % F);
    float v = fmaf(dinv[i], agg[t] + s[t], bias[j]);
    h[t] = (__bf16)fmaxf(v, 0.0f);
}

// ------------------------------------------- weight fragment pre-packing
// Per-lane WMMA-B bf16 fragment layout: lane = (hi<<4)|r, kbase = hi*8,
// frag[i] = W[k(kbase+i)][col], frag[8+i] = W[k(kbase+16+i)][col].

// W2 [64,32] -> 4 frags (f = ct*2 + chunk), each 32 lanes x 16 bf16
__global__ void tsp_packB2_kernel(const float* __restrict__ W2,
                                  __bf16* __restrict__ wsB2) {
    int t = threadIdx.x;                     // 128 threads
    if (t >= 128) return;
    int lane = t & 31, f = t >> 5;
    int ct = f >> 1, ch = f & 1;
    int r = lane & 15, kbase = (lane >> 4) * 8;
    __bf16* dst = wsB2 + t * 16;
#pragma unroll
    for (int i = 0; i < 8; ++i) {
        int kA = ch * 32 + kbase + i;
        int kB = ch * 32 + kbase + 16 + i;
        dst[i]     = (__bf16)W2[kA * 32 + ct * 16 + r];
        dst[8 + i] = (__bf16)W2[kB * 32 + ct * 16 + r];
    }
}

// Wm1 [66,16] zero-padded to K=96 -> 3 frags, each 32 lanes x 16 bf16
__global__ void tsp_packBm_kernel(const float* __restrict__ Wm1,
                                  __bf16* __restrict__ wsBm) {
    int t = threadIdx.x;                     // 96 threads
    if (t >= 96) return;
    int lane = t & 31, ch = t >> 5;
    int r = lane & 15, kbase = (lane >> 4) * 8;
    __bf16* dst = wsBm + t * 16;
#pragma unroll
    for (int i = 0; i < 8; ++i) {
        int kA = ch * 32 + kbase + i;
        int kB = ch * 32 + kbase + 16 + i;
        dst[i]     = (kA < 66) ? (__bf16)Wm1[kA * 16 + r] : (__bf16)0.0f;
        dst[8 + i] = (kB < 66) ? (__bf16)Wm1[kB * 16 + r] : (__bf16)0.0f;
    }
}

// ------------------------------------------- layer 2 GEMM: s2 = dinv*(h1@W2)
// One wave per 16-node tile; K=64 (2 bf16 chunks), 32 output cols (2 tiles).
__global__ __launch_bounds__(256) void tsp_l2_wmma_kernel(
    const __bf16* __restrict__ h1,   // [N,64] bf16
    const __bf16* __restrict__ wsB2, // packed fragments
    const float* __restrict__ dinv,  // [N]
    float* __restrict__ s2,          // [N,32]
    int N) {
    int wave = (blockIdx.x * blockDim.x + threadIdx.x) >> 5;
    int lane = threadIdx.x & 31;
    int m0 = wave * 16;
    if (m0 >= N) return;
    int r = lane & 15;
    int hi = lane >> 4;
    int kbase = hi * 8;

    const __bf16* hrow = h1 + (long long)(m0 + r) * 64;
    v8bf lo, hh;
    lo = *(const v8bf*)(hrow + kbase);
    hh = *(const v8bf*)(hrow + kbase + 16);
    v16bf a0 = CAT16(lo, hh);
    lo = *(const v8bf*)(hrow + 32 + kbase);
    hh = *(const v8bf*)(hrow + 32 + kbase + 16);
    v16bf a1 = CAT16(lo, hh);

    v16bf b00 = *(const v16bf*)(wsB2 + (0 * 32 + lane) * 16);  // ct0,ch0
    v16bf b01 = *(const v16bf*)(wsB2 + (1 * 32 + lane) * 16);  // ct0,ch1
    v16bf b10 = *(const v16bf*)(wsB2 + (2 * 32 + lane) * 16);  // ct1,ch0
    v16bf b11 = *(const v16bf*)(wsB2 + (3 * 32 + lane) * 16);  // ct1,ch1

    v8f c0 = {}, c1 = {};
    c0 = __builtin_amdgcn_wmma_f32_16x16x32_bf16(false, a0, false, b00, (short)0, c0, false, false);
    c0 = __builtin_amdgcn_wmma_f32_16x16x32_bf16(false, a1, false, b01, (short)0, c0, false, false);
    c1 = __builtin_amdgcn_wmma_f32_16x16x32_bf16(false, a0, false, b10, (short)0, c1, false, false);
    c1 = __builtin_amdgcn_wmma_f32_16x16x32_bf16(false, a1, false, b11, (short)0, c1, false, false);

#pragma unroll
    for (int j = 0; j < 8; ++j) {
        int m = m0 + j + 8 * hi;            // C/D layout: VGPR j -> M = j + 8*hi
        float dv = dinv[m];
        s2[(long long)m * 32 + r]      = dv * c0[j];
        s2[(long long)m * 32 + 16 + r] = dv * c1[j];
    }
}

// -------------------------------------------------------- fused edge MLP
// out[e] = (relu(concat(h2[row],h2[col],eattr) @ Wm1 + bm1)) @ Wm2 + bm2
// One wave per 16-edge tile; K padded 66 -> 96 (3 bf16 chunks, chunk2 sparse).
__global__ __launch_bounds__(256) void tsp_edge_mlp_kernel(
    const __bf16* __restrict__ h2,    // [N,32] bf16
    const long long* __restrict__ row,
    const long long* __restrict__ col,
    const float* __restrict__ eattr,  // [E,2]
    const __bf16* __restrict__ wsBm,  // packed Wm1 fragments
    const float* __restrict__ bm1,    // [16]
    const float* __restrict__ Wm2,    // [16]
    const float* __restrict__ bm2,    // [1]
    float* __restrict__ out,          // [E]
    int E) {
    int wave = (blockIdx.x * blockDim.x + threadIdx.x) >> 5;
    int lane = threadIdx.x & 31;
    int e0 = wave * 16;
    if (e0 >= E) return;
    int r = lane & 15;
    int hi = lane >> 4;
    int kbase = hi * 8;
    int e = e0 + r;
    int src = (int)row[e];
    int dst = (int)col[e];

    const __bf16* hr = h2 + (long long)src * 32;
    const __bf16* hc = h2 + (long long)dst * 32;
    v8bf lo, hh;
    lo = *(const v8bf*)(hr + kbase);
    hh = *(const v8bf*)(hr + kbase + 16);
    v16bf a0 = CAT16(lo, hh);
    lo = *(const v8bf*)(hc + kbase);
    hh = *(const v8bf*)(hc + kbase + 16);
    v16bf a1 = CAT16(lo, hh);

    float2 ea = ((const float2*)eattr)[e];
    v16bf a2;
#pragma unroll
    for (int i = 0; i < 16; ++i) a2[i] = (__bf16)0.0f;
    if (hi == 0) {  // chunk2: only global k = 64,65 live (K positions 0,1)
        a2[0] = (__bf16)ea.x;
        a2[1] = (__bf16)ea.y;
    }

    v16bf b0 = *(const v16bf*)(wsBm + (0 * 32 + lane) * 16);
    v16bf b1 = *(const v16bf*)(wsBm + (1 * 32 + lane) * 16);
    v16bf b2 = *(const v16bf*)(wsBm + (2 * 32 + lane) * 16);

    v8f c = {};
    c = __builtin_amdgcn_wmma_f32_16x16x32_bf16(false, a0, false, b0, (short)0, c, false, false);
    c = __builtin_amdgcn_wmma_f32_16x16x32_bf16(false, a1, false, b1, (short)0, c, false, false);
    c = __builtin_amdgcn_wmma_f32_16x16x32_bf16(false, a2, false, b2, (short)0, c, false, false);

    // hmid = relu(c + bm1[N]);  out[e] = sum_N hmid*Wm2[N] + bm2
    float bias = bm1[r];
    float w2n  = Wm2[r];
    float outb = bm2[0];
#pragma unroll
    for (int j = 0; j < 8; ++j) {
        float v = fmaxf(c[j] + bias, 0.0f) * w2n;
        v += __shfl_xor(v, 1, 32);   // reduce over N within each 16-lane half
        v += __shfl_xor(v, 2, 32);
        v += __shfl_xor(v, 4, 32);
        v += __shfl_xor(v, 8, 32);
        if (r == 0) out[e0 + j + 8 * hi] = v + outb;
    }
}

// ---------------------------------------------------------------- launcher

extern "C" void kernel_launch(void* const* d_in, const int* in_sizes, int n_in,
                              void* d_out, int out_size, void* d_ws, size_t ws_size,
                              hipStream_t stream) {
    const float*      x     = (const float*)d_in[0];
    const long long*  eidx  = (const long long*)d_in[1];
    const float*      eattr = (const float*)d_in[2];
    const float*      W1    = (const float*)d_in[3];
    const float*      b1    = (const float*)d_in[4];
    const float*      W2    = (const float*)d_in[5];
    const float*      b2    = (const float*)d_in[6];
    const float*      Wm1   = (const float*)d_in[7];
    const float*      bm1   = (const float*)d_in[8];
    const float*      Wm2   = (const float*)d_in[9];
    const float*      bm2   = (const float*)d_in[10];
    float*            out   = (float*)d_out;

    const int N = in_sizes[0] / 2;   // [N,2] node features
    const int E = in_sizes[1] / 2;   // [2,E] edge index
    const long long* row = eidx;
    const long long* col = eidx + E;

    // workspace layout
    float*  buf0 = (float*)d_ws;                     // N*64 f32: s1 -> (s2|agg2)
    float*  buf1 = buf0 + (long long)N * 64;         // N*64 f32: agg1
    float*  deg  = buf1 + (long long)N * 64;         // N f32: deg -> dinv
    __bf16* h1bf = (__bf16*)(deg + N);               // N*64 bf16
    __bf16* h2bf = h1bf + (long long)N * 64;         // N*32 bf16
    __bf16* wsB2 = h2bf + (long long)N * 32;         // 4*32*16 bf16
    __bf16* wsBm = wsB2 + 4 * 32 * 16;               // 3*32*16 bf16

    const int B = 256;
    auto blocks = [](long long n, int b) { return (unsigned)((n + b - 1) / b); };

    // one-time weight fragment packing (independent of node/edge passes)
    tsp_packB2_kernel<<<1, 128, 0, stream>>>(W2, wsB2);
    tsp_packBm_kernel<<<1, 96, 0, stream>>>(Wm1, wsBm);

    // degree + dinv
    tsp_fill_kernel<<<blocks(N, B), B, 0, stream>>>(deg, 1.0f, N);
    tsp_fill_kernel<<<blocks((long long)N * 64, B), B, 0, stream>>>(buf1, 0.0f, (long long)N * 64);
    tsp_deg_kernel<<<blocks(E, B), B, 0, stream>>>(col, deg, E);
    tsp_rsqrt_kernel<<<blocks(N, B), B, 0, stream>>>(deg, N);

    // layer 1: s1 = dinv*(x@W1); scatter; h1 = relu(dinv*(agg1+s1)+b1) -> bf16
    tsp_s1_kernel<<<blocks((long long)N * 64, B), B, 0, stream>>>(x, W1, deg, buf0, N);
    tsp_scatter_kernel<<<blocks((long long)E * 16, B), B, 0, stream>>>(buf0, row, col, buf1, E, 64);
    tsp_finalize_bf16_kernel<<<blocks((long long)N * 64, B), B, 0, stream>>>(buf0, buf1, deg, b1, h1bf, N, 64);

    // layer 2: s2 = dinv*(h1@W2) via WMMA; scatter; h2 = relu(...) -> bf16
    float* s2   = buf0;
    float* agg2 = buf0 + (long long)N * 32;
    tsp_fill_kernel<<<blocks((long long)N * 32, B), B, 0, stream>>>(agg2, 0.0f, (long long)N * 32);
    {
        long long waves = (N + 15) / 16;
        tsp_l2_wmma_kernel<<<blocks(waves * 32, B), B, 0, stream>>>(h1bf, wsB2, deg, s2, N);
    }
    tsp_scatter_kernel<<<blocks((long long)E * 8, B), B, 0, stream>>>(s2, row, col, agg2, E, 32);
    tsp_finalize_bf16_kernel<<<blocks((long long)N * 32, B), B, 0, stream>>>(s2, agg2, deg, b2, h2bf, N, 32);

    // fused edge MLP via WMMA
    {
        long long waves = (E + 15) / 16;
        tsp_edge_mlp_kernel<<<blocks(waves * 32, B), B, 0, stream>>>(
            h2bf, row, col, eattr, wsBm, bm1, Wm2, bm2, out, E);
    }
}
